// Skill_Evolve_Hetero_11055245820281
// MI455X (gfx1250) — compile-verified, matrix operand order checked by reading, not run on hardware
//
#include <hip/hip_runtime.h>
#include <hip/hip_bf16.h>
#include <math.h>

#define NN 10000
#define DD 128
#define TT 8
#define BB 1024
#define EE 80000
#define CC 80000

typedef __attribute__((ext_vector_type(16))) _Float16 v16h;
typedef __attribute__((ext_vector_type(8)))  float    v8f;

// ---------- helpers ----------
__device__ __forceinline__ unsigned enc_f32(float f) {
  unsigned u = __float_as_uint(f);
  return (u & 0x80000000u) ? ~u : (u | 0x80000000u);   // monotone float->uint
}
__device__ __forceinline__ float dec_f32(unsigned u) {
  return (u & 0x80000000u) ? __uint_as_float(u & 0x7fffffffu) : __uint_as_float(~u);
}
__device__ __forceinline__ float sigm(float x)  { return 1.f / (1.f + __expf(-x)); }
__device__ __forceinline__ float lrelu(float x, float s) { return x > 0.f ? x : s * x; }

// ---------- generic f16 WMMA GEMM (templated: no runtime branch in K loop) ----
// Out[m,n] = act( scale * ( (ACCUM?Out:0) + A[M,K] @ B + bias[n] ) )
// TRANSB==0 -> B is [K,N] row-major ; TRANSB==1 -> B is [N,K] row-major (A@B^T)
template <int TRANSB, int ACCUM>
__global__ __launch_bounds__(256) void gemm_wmma_f16(
    const float* __restrict__ A, int lda,
    const float* __restrict__ B,
    const float* __restrict__ bias,
    float* __restrict__ Out, int ldOut,
    int M, int K, int Nsz,
    int act, float scale)
{
  const int wave = threadIdx.x >> 5;
  const int lane = threadIdx.x & 31;
  const int ntile = blockIdx.x * (blockDim.x >> 5) + wave;
  const int mtile = blockIdx.y;
  const int n0 = ntile * 16, m0 = mtile * 16;
  if (n0 >= Nsz) return;                    // wave-uniform: EXEC stays all-ones

  const int hl  = lane >> 4;                // 0/1: lane half
  const int l15 = lane & 15;

  v8f c;
#pragma unroll
  for (int r = 0; r < 8; ++r)
    c[r] = ACCUM ? Out[(size_t)(m0 + 8 * hl + r) * ldOut + (n0 + l15)] : 0.0f;

  const int am = m0 + l15;                  // A row for this lane
  const int bn = n0 + l15;                  // B col for this lane
  for (int kk = 0; kk < K; kk += 32) {
    const int kb = kk + (hl << 3);          // K-base: +8 for upper lane half
    v16h a, b;
    const float* Ap = A + (size_t)am * lda + kb;
#pragma unroll
    for (int e = 0; e < 8; ++e) {           // e<8: K=kb+e ; e>=8: K=kb+16+e
      a[e]     = (_Float16)Ap[e];
      a[e + 8] = (_Float16)Ap[e + 16];
    }
    if (TRANSB) {
      const float* Bp = B + (size_t)bn * K + kb;
#pragma unroll
      for (int e = 0; e < 8; ++e) {
        b[e]     = (_Float16)Bp[e];
        b[e + 8] = (_Float16)Bp[e + 16];
      }
    } else {
      const float* Bp = B + (size_t)kb * Nsz + bn;
#pragma unroll
      for (int e = 0; e < 8; ++e) {
        b[e]     = (_Float16)Bp[(size_t)e * Nsz];
        b[e + 8] = (_Float16)Bp[(size_t)(e + 16) * Nsz];
      }
    }
    c = __builtin_amdgcn_wmma_f32_16x16x32_f16(false, a, false, b,
                                               (short)0, c, false, false);
  }

  const float bv = bias ? bias[n0 + l15] : 0.0f;
#pragma unroll
  for (int r = 0; r < 8; ++r) {
    float v = (c[r] + bv) * scale;
    if (act == 1) v = lrelu(v, 0.01f);
    Out[(size_t)(m0 + 8 * hl + r) * ldOut + (n0 + l15)] = v;
  }
}

// ---------- fragment pre-pack: f32 matrix -> f16 in WMMA register order -------
// frag[(tile*ktiles + kt)*32 + lane][e]  with
//   K index = kt*32 + 8*(lane>>4) + 16*(e>>3) + (e&7)
//   row/col = tile*16 + (lane&15)
// IS_A==1: Mtx is [rows, K] row-major (A operand); IS_A==0: Mtx is [K, cols] (B operand)
template <int IS_A>
__global__ void pack_frag_k(const float* __restrict__ Mtx, int ld,
                            _Float16* __restrict__ outF, int tiles, int ktiles) {
  int idx = blockIdx.x * blockDim.x + threadIdx.x;   // tiles*ktiles*32
  if (idx >= tiles * ktiles * 32) return;
  int lane = idx & 31;
  int kt   = (idx >> 5) % ktiles;
  int tile = (idx >> 5) / ktiles;
  int rc = tile * 16 + (lane & 15);
  int kb = kt * 32 + ((lane >> 4) << 3);
  v16h v;
#pragma unroll
  for (int e = 0; e < 8; ++e) {
    int k0 = kb + e, k1 = kb + 16 + e;
    float f0 = IS_A ? Mtx[(size_t)rc * ld + k0] : Mtx[(size_t)k0 * ld + rc];
    float f1 = IS_A ? Mtx[(size_t)rc * ld + k1] : Mtx[(size_t)k1 * ld + rc];
    v[e]     = (_Float16)f0;
    v[e + 8] = (_Float16)f1;
  }
  ((v16h*)outF)[idx] = v;
}

// ---------- decoder GEMM: pre-packed f16 fragments, K=128, fused row-max ------
// Out[M=1024, Nsz] = A(1024x128) @ B(128xNsz) + bias ; rowmax tracked for log_softmax
__global__ __launch_bounds__(256) void dec_gemm_k(
    const _Float16* __restrict__ Asw, const _Float16* __restrict__ Bsw,
    const float* __restrict__ bias, float* __restrict__ Out,
    unsigned* __restrict__ rowmax, int Nsz)
{
  const int wave = threadIdx.x >> 5;
  const int lane = threadIdx.x & 31;
  const int ntile = blockIdx.x * (blockDim.x >> 5) + wave;
  const int mtile = blockIdx.y;
  const int n0 = ntile * 16, m0 = mtile * 16;
  if (n0 >= Nsz) return;                    // wave-uniform

  const v16h* Af = (const v16h*)Asw;
  const v16h* Bf = (const v16h*)Bsw;
  const int hl  = lane >> 4;
  const int l15 = lane & 15;

  __builtin_prefetch(bias + n0, 0, 3);                       // -> global_prefetch_b8
  __builtin_prefetch(&Bf[((size_t)ntile * 4 + 2) * 32 + lane], 0, 0);

  v8f c;
#pragma unroll
  for (int r = 0; r < 8; ++r) c[r] = 0.0f;
#pragma unroll
  for (int kt = 0; kt < 4; ++kt) {          // K = 4 x 32
    v16h a = Af[(size_t)(mtile * 4 + kt) * 32 + lane];
    v16h b = Bf[((size_t)ntile * 4 + kt) * 32 + lane];
    c = __builtin_amdgcn_wmma_f32_16x16x32_f16(false, a, false, b,
                                               (short)0, c, false, false);
  }

  const float bv = bias[n0 + l15];
#pragma unroll
  for (int r = 0; r < 8; ++r) c[r] += bv;
#pragma unroll
  for (int r = 0; r < 8; ++r) {             // row-max over the 16 N-lanes
    float v = c[r];
    for (int off = 1; off < 16; off <<= 1)
      v = fmaxf(v, __shfl_xor(v, off, 32));
    if (l15 == 0) atomicMax(&rowmax[m0 + 8 * hl + r], enc_f32(v));
  }
#pragma unroll
  for (int r = 0; r < 8; ++r)
    Out[(size_t)(m0 + 8 * hl + r) * Nsz + (n0 + l15)] = c[r];
}

// ---------- small utility kernels ----------
__global__ void fill_f32(float* p, int n, float v) {
  int i = blockIdx.x * blockDim.x + threadIdx.x;
  if (i < n) p[i] = v;
}
__global__ void fill_u32(unsigned* p, int n, unsigned v) {
  int i = blockIdx.x * blockDim.x + threadIdx.x;
  if (i < n) p[i] = v;
}

// per-node attention scalars: h@a_src, h@a_dst for both GATs
__global__ void alpha_k(const float* __restrict__ hp, const float* __restrict__ hc,
                        const float* __restrict__ asp, const float* __restrict__ adp,
                        const float* __restrict__ asc, const float* __restrict__ adc,
                        float* aps, float* apd, float* acs, float* acd) {
  int n = blockIdx.x * blockDim.x + threadIdx.x;
  if (n >= NN) return;
  const float* p1 = hp + (size_t)n * DD;
  const float* p2 = hc + (size_t)n * DD;
  float s1 = 0, s2 = 0, s3 = 0, s4 = 0;
  for (int j = 0; j < DD; ++j) {
    s1 += p1[j] * asp[j]; s2 += p1[j] * adp[j];
    s3 += p2[j] * asc[j]; s4 += p2[j] * adc[j];
  }
  aps[n] = s1; apd[n] = s2; acs[n] = s3; acd[n] = s4;
}

__global__ void gat_logit_k(const int* __restrict__ src, const int* __restrict__ dst,
                            const float* __restrict__ avs, const float* __restrict__ avd,
                            float* logit, unsigned* mx) {
  int e = blockIdx.x * blockDim.x + threadIdx.x;
  if (e >= EE) return;
  float l = lrelu(avs[src[e]] + avd[dst[e]], 0.2f);
  logit[e] = l;
  atomicMax(&mx[dst[e]], enc_f32(l));
}

__global__ void gat_exp_k(const int* __restrict__ dst, const float* __restrict__ logit,
                          const unsigned* __restrict__ mx, float* exb, float* den) {
  int e = blockIdx.x * blockDim.x + threadIdx.x;
  if (e >= EE) return;
  int d = dst[e];
  float ex = __expf(logit[e] - dec_f32(mx[d]));
  exb[e] = ex;
  atomicAdd(&den[d], ex);
}

__global__ void gat_scatter_k(const int* __restrict__ src, const int* __restrict__ dst,
                              const float* __restrict__ h, const float* __restrict__ exb,
                              const float* __restrict__ den, float* acc) {
  int idx = blockIdx.x * blockDim.x + threadIdx.x;   // EE * 32
  if (idx >= EE * 32) return;
  int e = idx >> 5, q = idx & 31;
  int s = src[e], d = dst[e];
  float coef = exb[e] / (den[d] + 1e-16f);
  float4 hv = ((const float4*)(h + (size_t)s * DD))[q];
  float* a = acc + (size_t)d * DD + q * 4;
  atomicAdd(a + 0, hv.x * coef); atomicAdd(a + 1, hv.y * coef);
  atomicAdd(a + 2, hv.z * coef); atomicAdd(a + 3, hv.w * coef);
}

__global__ void sage_scatter_k(const int* __restrict__ src, const int* __restrict__ dst,
                               const float* __restrict__ x, float* ssum, float* cnt) {
  int idx = blockIdx.x * blockDim.x + threadIdx.x;   // EE * 32
  if (idx >= EE * 32) return;
  int e = idx >> 5, q = idx & 31;
  int s = src[e], d = dst[e];
  float4 xv = ((const float4*)(x + (size_t)s * DD))[q];
  float* a = ssum + (size_t)d * DD + q * 4;
  atomicAdd(a + 0, xv.x); atomicAdd(a + 1, xv.y);
  atomicAdd(a + 2, xv.z); atomicAdd(a + 3, xv.w);
  if (q == 0) atomicAdd(&cnt[d], 1.0f);
}

// gather GAT outputs + biases + x@Wsr at s_nodes; base goes into s_bt slice
__global__ void gather_combine_k(const int* __restrict__ s_nodes,
                                 const float* __restrict__ accp, const float* __restrict__ accc,
                                 const float* __restrict__ ssum, const float* __restrict__ cnt,
                                 const float* __restrict__ hsr,
                                 const float* __restrict__ bp, const float* __restrict__ bc,
                                 const float* __restrict__ bs,
                                 float* meanB, float* s_bt, int t) {
  int idx = blockIdx.x * blockDim.x + threadIdx.x;   // BB * 32
  if (idx >= BB * 32) return;
  int b = idx >> 5, q = idx & 31;
  int s = s_nodes[b];
  float4 ap = ((const float4*)(accp + (size_t)s * DD))[q];
  float4 ac = ((const float4*)(accc + (size_t)s * DD))[q];
  float4 hs = ((const float4*)(hsr  + (size_t)s * DD))[q];
  float4 sm = ((const float4*)(ssum + (size_t)s * DD))[q];
  float4 vbp = ((const float4*)bp)[q];
  float4 vbc = ((const float4*)bc)[q];
  float4 vbs = ((const float4*)bs)[q];
  float cn = fmaxf(cnt[s], 1.0f);
  float4 base;
  base.x = ap.x + vbp.x + ac.x + vbc.x + hs.x + vbs.x;
  base.y = ap.y + vbp.y + ac.y + vbc.y + hs.y + vbs.y;
  base.z = ap.z + vbp.z + ac.z + vbc.z + hs.z + vbs.z;
  base.w = ap.w + vbp.w + ac.w + vbc.w + hs.w + vbs.w;
  ((float4*)(s_bt + (size_t)b * TT * DD + (size_t)t * DD))[q] = base;
  float4 mn; mn.x = sm.x / cn; mn.y = sm.y / cn; mn.z = sm.z / cn; mn.w = sm.w / cn;
  ((float4*)(meanB + (size_t)b * DD))[q] = mn;
}

// Z = [ lrelu(x*w+b) | s_bt ]  rows (b,t), 2D cols
__global__ void build_zamp_k(const float* __restrict__ x, const float* __restrict__ w,
                             const float* __restrict__ bb, const float* __restrict__ s_bt,
                             float* Z) {
  int idx = blockIdx.x * blockDim.x + threadIdx.x;   // BB*TT*2DD
  if (idx >= BB * TT * 2 * DD) return;
  int r = idx >> 8, col = idx & 255;                 // 2*DD == 256
  float v;
  if (col < DD) v = lrelu(x[r] * w[col] + bb[col], 0.01f);
  else          v = s_bt[(size_t)r * DD + (col - DD)];
  Z[idx] = v;
}

// add sinusoidal positional encoding to both amplified sequences
__global__ void pe_add_k(float* amp_d, float* amp_s) {
  int idx = blockIdx.x * blockDim.x + threadIdx.x;   // BB*TT*DD
  if (idx >= BB * TT * DD) return;
  int j = idx & (DD - 1);
  int t = (idx >> 7) & (TT - 1);
  int jj = j & ~1;
  float div = __expf(-(float)jj * (9.210340372f / (float)DD));   // ln(10000)
  float ang = (float)t * div;
  float pe = (j & 1) ? __cosf(ang) : __sinf(ang);
  amp_d[idx] += pe;
  amp_s[idx] += pe;
}

// LSTM pointwise gates; captures h at the last valid step into enc
__global__ void lstm_gate_k(const float* __restrict__ G, float* h, float* c, float* enc,
                            const int* __restrict__ lv, const int* __restrict__ tsv, int t) {
  int idx = blockIdx.x * blockDim.x + threadIdx.x;   // BB*DD
  if (idx >= BB * DD) return;
  int b = idx >> 7, j = idx & (DD - 1);
  const float* g = G + (size_t)b * 4 * DD;
  float ig = sigm(g[j]);
  float fg = sigm(g[DD + j]);
  float gg = tanhf(g[2 * DD + j]);
  float og = sigm(g[3 * DD + j]);
  float cn = fg * c[idx] + ig * gg;
  float hn = og * tanhf(cn);
  c[idx] = cn; h[idx] = hn;
  int len = tsv[b] + lv[b];
  len = len < 1 ? 1 : (len > TT ? TT : len);
  if (t == len - 1) enc[idx] = hn;
}

__global__ void build_zm_k(const float* __restrict__ ed, const float* __restrict__ es,
                           const float* __restrict__ s_bt, const int* __restrict__ te,
                           float* Zm) {
  int idx = blockIdx.x * blockDim.x + threadIdx.x;   // BB*3DD
  if (idx >= BB * 3 * DD) return;
  int b = idx / (3 * DD), col = idx % (3 * DD);
  float v;
  if (col < DD)          v = ed[(size_t)b * DD + col];
  else if (col < 2 * DD) v = es[(size_t)b * DD + (col - DD)];
  else v = s_bt[(size_t)b * TT * DD + (size_t)te[b] * DD + (col - 2 * DD)];
  Zm[idx] = v;
}

__global__ void build_za_k(const float* __restrict__ ex, const float* __restrict__ mg,
                           float* Za) {
  int idx = blockIdx.x * blockDim.x + threadIdx.x;   // BB*2DD
  if (idx >= BB * 2 * DD) return;
  int b = idx >> 8, col = idx & 255;
  Za[idx] = (col < DD) ? ex[(size_t)b * DD + col] : mg[(size_t)b * DD + (col - DD)];
}

// log-sum-exp per output row (2*BB rows of CC logits)
__global__ void lse_k(const float* __restrict__ logits, const unsigned* __restrict__ rowmax,
                      float* lse) {
  __shared__ float red[256];
  int row = blockIdx.x;
  float m = dec_f32(rowmax[row]);
  const float* p = logits + (size_t)row * CC;
  float s = 0.f;
  for (int i = threadIdx.x; i < CC; i += blockDim.x) s += __expf(p[i] - m);
  red[threadIdx.x] = s;
  __syncthreads();
  for (int st = 128; st > 0; st >>= 1) {
    if ((int)threadIdx.x < st) red[threadIdx.x] += red[threadIdx.x + st];
    __syncthreads();
  }
  if (threadIdx.x == 0) lse[row] = m + __logf(red[0]);
}

__global__ void lsm_fin_k(float* out, const float* __restrict__ lse) {
  int i = blockIdx.x * blockDim.x + threadIdx.x;     // (2*BB*CC)/4 float4s
  if (i >= (2 * BB * (CC / 4))) return;
  int row = i / (CC / 4);
  float l = lse[row];
  float4* p = ((float4*)out) + i;
  float4 v = *p;
  v.x -= l; v.y -= l; v.z -= l; v.w -= l;
  *p = v;
}

// ---------- host orchestration ----------
extern "C" void kernel_launch(void* const* d_in, const int* in_sizes, int n_in,
                              void* d_out, int out_size, void* d_ws, size_t ws_size,
                              hipStream_t stream) {
  (void)in_sizes; (void)n_in; (void)out_size; (void)ws_size;
  const float* d_x     = (const float*)d_in[0];
  const float* s_x     = (const float*)d_in[1];
  const int*   lv      = (const int*)d_in[2];
  const int*   t_s     = (const int*)d_in[3];
  const int*   s_nodes = (const int*)d_in[4];
  const int*   t_e     = (const int*)d_in[5];
  const int*   edges   = (const int*)d_in[6];
  const float* emb     = (const float*)d_in[7];
  const float* Wp  = (const float*)d_in[8];
  const float* asp = (const float*)d_in[9];
  const float* adp = (const float*)d_in[10];
  const float* bp  = (const float*)d_in[11];
  const float* Wc  = (const float*)d_in[12];
  const float* asc = (const float*)d_in[13];
  const float* adc = (const float*)d_in[14];
  const float* bc  = (const float*)d_in[15];
  const float* Wsl = (const float*)d_in[16];
  const float* Wsr = (const float*)d_in[17];
  const float* bs  = (const float*)d_in[18];
  const float* w_amp_d = (const float*)d_in[19];
  const float* b_amp_d = (const float*)d_in[20];
  const float* w_amp_s = (const float*)d_in[21];
  const float* b_amp_s = (const float*)d_in[22];
  const float* W_skamp = (const float*)d_in[23];
  const float* b_skamp = (const float*)d_in[24];
  const float* Wih_d = (const float*)d_in[25];
  const float* Whh_d = (const float*)d_in[26];
  const float* bih_d = (const float*)d_in[27];
  const float* bhh_d = (const float*)d_in[28];
  const float* Wih_s = (const float*)d_in[29];
  const float* Whh_s = (const float*)d_in[30];
  const float* bih_s = (const float*)d_in[31];
  const float* bhh_s = (const float*)d_in[32];
  const float* Wm  = (const float*)d_in[33];
  const float* bm  = (const float*)d_in[34];
  const float* Wa_d = (const float*)d_in[35];
  const float* Wa_s = (const float*)d_in[36];
  const float* Wdec_d = (const float*)d_in[37];
  const float* bdec_d = (const float*)d_in[38];
  const float* Wdec_s = (const float*)d_in[39];
  const float* bdec_s = (const float*)d_in[40];
  float* out = (float*)d_out;

  // workspace carve-out (float units; every size is a multiple of 8 -> 32B alignment)
  float* ws = (float*)d_ws;
  size_t off = 0;
  auto alloc = [&](size_t n) { float* p = ws + off; off += n; return p; };
  float* hp   = alloc((size_t)NN * DD);
  float* hc   = alloc((size_t)NN * DD);
  float* hsr  = alloc((size_t)NN * DD);
  float* accp = alloc((size_t)NN * DD);       // accp, accc, ssum contiguous
  float* accc = alloc((size_t)NN * DD);
  float* ssum = alloc((size_t)NN * DD);
  float* aps = alloc(NN);
  float* apd = alloc(NN);
  float* acs = alloc(NN);
  float* acd = alloc(NN);
  float* denp = alloc(NN);                    // denp, denc, cnt contiguous
  float* denc = alloc(NN);
  float* cnt  = alloc(NN);
  float* logit = alloc(EE);
  float* exb   = alloc(EE);
  float* meanB = alloc((size_t)BB * DD);
  float* s_bt  = alloc((size_t)BB * TT * DD);
  float* Zamp  = alloc((size_t)BB * TT * 2 * DD);
  float* amp_d = alloc((size_t)BB * TT * DD);
  float* amp_s = alloc((size_t)BB * TT * DD);
  float* G     = alloc((size_t)BB * 4 * DD);
  float* hbuf  = alloc((size_t)BB * DD);      // hbuf, cbuf contiguous
  float* cbuf  = alloc((size_t)BB * DD);
  float* enc_d = alloc((size_t)BB * DD);
  float* enc_s = alloc((size_t)BB * DD);
  float* Zm    = alloc((size_t)BB * 3 * DD);
  float* merged= alloc((size_t)BB * DD);
  float* Za    = alloc((size_t)BB * 2 * DD);
  float* dh    = alloc((size_t)BB * DD);
  float* sh    = alloc((size_t)BB * DD);
  float* lse   = alloc(2 * BB);
  unsigned* maxp   = (unsigned*)alloc(NN);    // maxp, maxc contiguous
  unsigned* maxc   = (unsigned*)alloc(NN);
  unsigned* rowmax = (unsigned*)alloc(2 * BB);
  // f16 fragment buffers (sizes in float units = f16 count / 2)
  const int NT_C = CC / 16;                   // 5000 n-tiles
  _Float16* AswD = (_Float16*)alloc((size_t)64 * 4 * 32 * 16 / 2);
  _Float16* AswS = (_Float16*)alloc((size_t)64 * 4 * 32 * 16 / 2);
  _Float16* BswD = (_Float16*)alloc((size_t)NT_C * 4 * 32 * 16 / 2);
  _Float16* BswS = (_Float16*)alloc((size_t)NT_C * 4 * 32 * 16 / 2);

  auto gemm = [&](const float* A, int lda, const float* Bm, int transB, const float* bias,
                  float* Out, int ldOut, int M, int K, int Nsz,
                  int accum, int act, float scale) {
    dim3 block(256);
    dim3 grid((Nsz / 16 + 7) / 8, M / 16);
    if (transB) {
      if (accum) gemm_wmma_f16<1, 1><<<grid, block, 0, stream>>>(A, lda, Bm, bias, Out, ldOut, M, K, Nsz, act, scale);
      else       gemm_wmma_f16<1, 0><<<grid, block, 0, stream>>>(A, lda, Bm, bias, Out, ldOut, M, K, Nsz, act, scale);
    } else {
      if (accum) gemm_wmma_f16<0, 1><<<grid, block, 0, stream>>>(A, lda, Bm, bias, Out, ldOut, M, K, Nsz, act, scale);
      else       gemm_wmma_f16<0, 0><<<grid, block, 0, stream>>>(A, lda, Bm, bias, Out, ldOut, M, K, Nsz, act, scale);
    }
  };
#define GRID1(n) dim3(((n) + 255) / 256), dim3(256), 0, stream

  // ---- stage A: time-invariant node transforms (WMMA) + attention scalars ----
  gemm(emb, DD, Wp,  0, nullptr, hp,  DD, NN, DD, DD, 0, 0, 1.f);
  gemm(emb, DD, Wc,  0, nullptr, hc,  DD, NN, DD, DD, 0, 0, 1.f);
  gemm(emb, DD, Wsr, 0, nullptr, hsr, DD, NN, DD, DD, 0, 0, 1.f);
  alpha_k<<<GRID1(NN)>>>(hp, hc, asp, adp, asc, adc, aps, apd, acs, acd);

  // ---- pre-pack decoder weights to f16 WMMA fragment order (L2-resident) ----
  pack_frag_k<0><<<GRID1(NT_C * 4 * 32)>>>(Wdec_d, CC, BswD, NT_C, 4);
  pack_frag_k<0><<<GRID1(NT_C * 4 * 32)>>>(Wdec_s, CC, BswS, NT_C, 4);

  // ---- per-time-step graph convolutions ----
  for (int t = 0; t < TT; ++t) {
    fill_f32<<<GRID1(3 * NN * DD)>>>(accp, 3 * NN * DD, 0.f);
    fill_f32<<<GRID1(3 * NN)>>>(denp, 3 * NN, 0.f);
    fill_u32<<<GRID1(2 * NN)>>>(maxp, 2 * NN, 0u);
    for (int conv = 0; conv < 2; ++conv) {       // two GATs
      const int* src = edges + (size_t)((t * 3 + conv) * 2 + 0) * EE;
      const int* dst = src + EE;
      const float* avs = conv ? acs : aps;
      const float* avd = conv ? acd : apd;
      unsigned* mx = conv ? maxc : maxp;
      float* den   = conv ? denc : denp;
      float* acc   = conv ? accc : accp;
      const float* hmat = conv ? hc : hp;
      gat_logit_k  <<<GRID1(EE)>>>(src, dst, avs, avd, logit, mx);
      gat_exp_k    <<<GRID1(EE)>>>(dst, logit, mx, exb, den);
      gat_scatter_k<<<GRID1(EE * 32)>>>(src, dst, hmat, exb, den, acc);
    }
    const int* src2 = edges + (size_t)((t * 3 + 2) * 2 + 0) * EE;  // SAGE
    const int* dst2 = src2 + EE;
    sage_scatter_k<<<GRID1(EE * 32)>>>(src2, dst2, emb, ssum, cnt);
    gather_combine_k<<<GRID1(BB * 32)>>>(s_nodes, accp, accc, ssum, cnt, hsr,
                                         bp, bc, bs, meanB, s_bt, t);
    // s_bt[:,t,:] = (base + meanB@Wsl) / 3   (WMMA, accumulate, scale)
    gemm(meanB, DD, Wsl, 0, nullptr, s_bt + (size_t)t * DD, TT * DD,
         BB, DD, DD, 1, 0, 1.f / 3.f);
  }

  // ---- amplify (WMMA) + positional encoding ----
  build_zamp_k<<<GRID1(BB * TT * 2 * DD)>>>(d_x, w_amp_d, b_amp_d, s_bt, Zamp);
  gemm(Zamp, 2 * DD, W_skamp, 0, b_skamp, amp_d, DD, BB * TT, 2 * DD, DD, 0, 1, 1.f);
  build_zamp_k<<<GRID1(BB * TT * 2 * DD)>>>(s_x, w_amp_s, b_amp_s, s_bt, Zamp);
  gemm(Zamp, 2 * DD, W_skamp, 0, b_skamp, amp_s, DD, BB * TT, 2 * DD, DD, 0, 1, 1.f);
  pe_add_k<<<GRID1(BB * TT * DD)>>>(amp_d, amp_s);

  // ---- two LSTMs (gate GEMMs via WMMA with transB) ----
  for (int which = 0; which < 2; ++which) {
    const float* amp = which ? amp_s : amp_d;
    const float* Wih = which ? Wih_s : Wih_d;
    const float* Whh = which ? Whh_s : Whh_d;
    const float* bih = which ? bih_s : bih_d;
    const float* bhh = which ? bhh_s : bhh_d;
    float* enc = which ? enc_s : enc_d;
    fill_f32<<<GRID1(2 * BB * DD)>>>(hbuf, 2 * BB * DD, 0.f);
    for (int t = 0; t < TT; ++t) {
      gemm(amp + (size_t)t * DD, TT * DD, Wih, 1, bih, G, 4 * DD,
           BB, DD, 4 * DD, 0, 0, 1.f);
      gemm(hbuf, DD, Whh, 1, bhh, G, 4 * DD, BB, DD, 4 * DD, 1, 0, 1.f);
      lstm_gate_k<<<GRID1(BB * DD)>>>(G, hbuf, cbuf, enc, lv, t_s, t);
    }
  }

  // ---- decode (WMMA) ----
  build_zm_k<<<GRID1(BB * 3 * DD)>>>(enc_d, enc_s, s_bt, t_e, Zm);
  gemm(Zm, 3 * DD, Wm, 0, bm, merged, DD, BB, 3 * DD, DD, 0, 1, 1.f);
  build_za_k<<<GRID1(BB * 2 * DD)>>>(enc_d, merged, Za);
  gemm(Za, 2 * DD, Wa_d, 0, nullptr, dh, DD, BB, 2 * DD, DD, 0, 1, 1.f);
  build_za_k<<<GRID1(BB * 2 * DD)>>>(enc_s, merged, Za);
  gemm(Za, 2 * DD, Wa_s, 0, nullptr, sh, DD, BB, 2 * DD, DD, 0, 1, 1.f);

  // ---- decoder GEMMs: pack A to f16 fragments, then f16-WMMA + fused row-max ----
  pack_frag_k<1><<<GRID1(64 * 4 * 32)>>>(dh, DD, AswD, 64, 4);
  pack_frag_k<1><<<GRID1(64 * 4 * 32)>>>(sh, DD, AswS, 64, 4);
  fill_u32<<<GRID1(2 * BB)>>>(rowmax, 2 * BB, 0u);
  {
    dim3 block(256);
    dim3 grid((NT_C + 7) / 8, BB / 16);
    dec_gemm_k<<<grid, block, 0, stream>>>(AswD, BswD, bdec_d, out,                   rowmax,      CC);
    dec_gemm_k<<<grid, block, 0, stream>>>(AswS, BswS, bdec_s, out + (size_t)BB * CC, rowmax + BB, CC);
  }

  // ---- log_softmax: LSE pass + vectorized subtract ----
  lse_k<<<dim3(2 * BB), dim3(256), 0, stream>>>(out, rowmax, lse);
  lsm_fin_k<<<GRID1(2 * BB * (CC / 4))>>>(out, lse);
#undef GRID1
}
